// ReLAttention_48043504173208
// MI455X (gfx1250) — matrix operand order repaired and checked by reading, not compiled, not run on hardware
//
#include <hip/hip_runtime.h>
#include <hip/hip_bf16.h>

// Problem constants (fixed by the reference)
constexpr int EMB    = 1024;
constexpr int NHEAD  = 16;
constexpr int HDIM   = 64;
constexpr int SEQLEN = 2048;
constexpr int NBATCH = 2;
constexpr int NTOK   = SEQLEN * NBATCH;   // 4096 token rows
constexpr float EPS  = 1e-8f;

typedef __attribute__((ext_vector_type(16))) __bf16 v16bf;
typedef __attribute__((ext_vector_type(8)))  __bf16 v8bf;
typedef __attribute__((ext_vector_type(4)))  __bf16 v4bf;
typedef __attribute__((ext_vector_type(8)))  float  v8f;

static __device__ __forceinline__ v16bf cat8(v8bf lo, v8bf hi) {
  return __builtin_shufflevector(lo, hi, 0,1,2,3,4,5,6,7,8,9,10,11,12,13,14,15);
}
static __device__ __forceinline__ v8f zero8() {
  v8f z = {0.f,0.f,0.f,0.f,0.f,0.f,0.f,0.f};
  return z;
}
// relu in one VALU op: v_med3_f32(x, 0, +inf)
static __device__ __forceinline__ float relu1(float x) {
  return __builtin_amdgcn_fmed3f(x, 0.f, __builtin_inff());
}

// ---------------------------------------------------------------------------
// Weight pre-pass: W [N][K] f32  ->  Bt [K][N] bf16  (done once per weight)
// ---------------------------------------------------------------------------
__global__ __launch_bounds__(256) void wtrans_bf16(
    const float* __restrict__ W, __bf16* __restrict__ Bt)
{
  __shared__ float tile[32][33];
  const int bx = blockIdx.x;       // K tile
  const int by = blockIdx.y;       // N tile
  const int tx = threadIdx.x & 31, ty = threadIdx.x >> 5;
  #pragma unroll
  for (int r = ty; r < 32; r += 8)
    tile[r][tx] = W[(size_t)(by * 32 + r) * EMB + bx * 32 + tx];
  __syncthreads();
  #pragma unroll
  for (int r = ty; r < 32; r += 8)
    Bt[(size_t)(bx * 32 + r) * EMB + by * 32 + tx] = (__bf16)tile[tx][r];
}

// ---------------------------------------------------------------------------
// GEMM: C[4096,1024] = epi( A[4096,1024] * Bt[1024,1024(kxn)] + bias )
// A is f32 (ABF16=false) or bf16 (ABF16=true); Bt is pre-transposed bf16.
// MODE 0: bf16 store C[row*N+col] (with scale)     (Q)
// MODE 1: bf16 store per-head transposed [d][s]    (K, V)
// MODE 2: f32  store C[row*N+col]                  (final -> d_out)
// BK=64: 16 WMMAs per barrier pair; register double-buffered global loads.
// ---------------------------------------------------------------------------
constexpr int BM = 128, BN = 128, BK = 64;
constexpr int AST = 72;    // Al row stride (bf16): 144B, 16B aligned, skewed
constexpr int BST = 136;   // Bl row stride (bf16): 272B, 16B aligned

template<bool ABF16, int MODE>
__global__ __launch_bounds__(256) void gemm_wmma(
    const void* __restrict__ Av, const __bf16* __restrict__ Bt,
    const float* __restrict__ bias, void* __restrict__ Cout, float scale)
{
  __shared__ __attribute__((aligned(16))) __bf16 Al[BM * AST];
  __shared__ __attribute__((aligned(16))) __bf16 Bl[BK * BST];   // [k][n]

  const int t    = threadIdx.x;
  const int lane = t & 31;
  const int w    = t >> 5;
  const int rowBase = blockIdx.y * BM;
  const int colBase = blockIdx.x * BN;

  const float*  Af = (const float*)Av;
  const __bf16* Ab = (const __bf16*)Av;

  v8f acc[8];
  #pragma unroll
  for (int j = 0; j < 8; ++j) acc[j] = zero8();

  const int ar   = lane & 15;
  const int koff = (lane >= 16) ? 8 : 0;

  // ---- register double-buffering ------------------------------------------
  float4 pa_f[8];           // ABF16=false: 128x64 f32 tile, 8 chunks/thread
  v8bf   pa_b[4];           // ABF16=true : 128x64 bf16 tile, 4 chunks/thread
  v8bf   pb[4];             // B tile 64x128 bf16, 4 chunks/thread

  auto prefetch = [&](int k0) {
    if (ABF16) {
      #pragma unroll
      for (int j = 0; j < 4; ++j) {
        int i = t + j * 256, r = i >> 3, ch = i & 7;
        pa_b[j] = *(const v8bf*)(Ab + (size_t)(rowBase + r) * EMB + k0 + ch * 8);
      }
    } else {
      #pragma unroll
      for (int j = 0; j < 8; ++j) {
        int i = t + j * 256, r = i >> 4, ch = i & 15;
        pa_f[j] = *(const float4*)(Af + (size_t)(rowBase + r) * EMB + k0 + ch * 4);
      }
    }
    #pragma unroll
    for (int j = 0; j < 4; ++j) {
      int i = t + j * 256, kr = i >> 4, ch = i & 15;
      pb[j] = *(const v8bf*)(Bt + (size_t)(k0 + kr) * EMB + colBase + ch * 8);
    }
  };
  auto stage = [&]() {
    if (ABF16) {
      #pragma unroll
      for (int j = 0; j < 4; ++j) {
        int i = t + j * 256, r = i >> 3, ch = i & 7;
        *(v8bf*)(Al + r * AST + ch * 8) = pa_b[j];
      }
    } else {
      #pragma unroll
      for (int j = 0; j < 8; ++j) {
        int i = t + j * 256, r = i >> 4, ch = i & 15;
        float4 g = pa_f[j];
        v4bf c = {(__bf16)g.x, (__bf16)g.y, (__bf16)g.z, (__bf16)g.w};
        *(v4bf*)(Al + r * AST + ch * 4) = c;
      }
    }
    #pragma unroll
    for (int j = 0; j < 4; ++j) {
      int i = t + j * 256, kr = i >> 4, ch = i & 15;
      *(v8bf*)(Bl + kr * BST + ch * 8) = pb[j];
    }
  };

  prefetch(0);
  for (int k0 = 0; k0 < EMB; k0 += BK) {
    __syncthreads();          // previous compute done reading LDS
    stage();
    __syncthreads();
    if (k0 + BK < EMB) prefetch(k0 + BK);   // overlaps with 16 WMMAs below

    #pragma unroll
    for (int ks = 0; ks < 2; ++ks) {
      const __bf16* arow = Al + (w * 16 + ar) * AST + ks * 32;
      v16bf afrag = cat8(*(const v8bf*)(arow + koff),
                         *(const v8bf*)(arow + 16 + koff));
      #pragma unroll
      for (int j = 0; j < 8; ++j) {
        const __bf16* brow = Bl + (ks * 32 + koff) * BST + j * 16 + ar;
        v16bf bfrag;
        #pragma unroll
        for (int s = 0; s < 16; ++s) bfrag[s] = brow[((s & 7) + (s >> 3) * 16) * BST];
        acc[j] = __builtin_amdgcn_wmma_f32_16x16x32_bf16(
            false, afrag, false, bfrag, (short)0, acc[j], false, false);
      }
    }
  }

  // Epilogue. C/D layout: VGPR r -> row r + 8*(lane>=16), col = lane&15
  const int cc = lane & 15;
  const int rr = (lane >> 4) * 8;
  #pragma unroll
  for (int j = 0; j < 8; ++j) {
    int col = colBase + j * 16 + cc;
    float bb = bias[col];
    #pragma unroll
    for (int r = 0; r < 8; ++r) {
      int row = rowBase + w * 16 + rr + r;
      float val = (acc[j][r] + bb) * scale;
      if (MODE == 0) {
        ((__bf16*)Cout)[(size_t)row * EMB + col] = (__bf16)val;
      } else if (MODE == 1) {
        int l_ = row >> 1, b_ = row & 1;       // token row = l*NBATCH + b
        int h_ = col >> 6, d_ = col & 63;
        ((__bf16*)Cout)[((size_t)((b_ * NHEAD + h_) * HDIM + d_)) * SEQLEN + l_] =
            (__bf16)val;
      } else {
        ((float*)Cout)[(size_t)row * EMB + col] = val;
      }
    }
  }
}

// ---------------------------------------------------------------------------
// Fused ReLU attention. Block = (b, h, 128 q-rows); 8 waves x 16 rows.
// K/V tiles streamed by CDNA5 async-to-LDS (global_load_async_to_lds_b128,
// ASYNCcnt) into double-buffered LDS -> one block barrier per s-tile.
// Stage 1: P = relu(Q K^T) spilled transposed (Pt[s][l], 1 ds_store_b128/tile).
// Stage 2: out^T = V^T * P^T (Pt is directly a valid B fragment).
// Never materializes the 2048x2048 score matrix (~512MB HBM saved).
// ---------------------------------------------------------------------------
constexpr int SB   = 64;   // s-tile width
constexpr int KST  = 72;   // Kl[d][s] stride (144B: aligned, bank-skewed)
constexpr int VTST = 72;   // Vl[d][s] stride
constexpr int PTST = 24;   // Pt[s][l] stride (48B: aligned, conflict-free)

__global__ __launch_bounds__(256) void attn_relu_wmma(
    const __bf16* __restrict__ qb, const __bf16* __restrict__ kT,
    const __bf16* __restrict__ vT, float* __restrict__ attn)
{
  __shared__ __attribute__((aligned(16))) __bf16 Kl[2 * 64 * KST];   // dbl-buf
  __shared__ __attribute__((aligned(16))) __bf16 Vl[2 * 64 * VTST];  // dbl-buf
  __shared__ __attribute__((aligned(16))) __bf16 Pt[8 * 64 * PTST];  // P^T/wave

  const int t = threadIdx.x, lane = t & 31, w = t >> 5;
  const int l0 = blockIdx.x * 128;
  const int bh = blockIdx.y;
  const int b = bh / NHEAD, h = bh % NHEAD;

  const __bf16* kbase = kT + (size_t)(b * NHEAD + h) * HDIM * SEQLEN;
  const __bf16* vbase = vT + (size_t)(b * NHEAD + h) * HDIM * SEQLEN;

  // staging geometry: thread -> (row = t>>3 [+32], chunk = t&7)
  const int srow = t >> 3, sch = t & 7;

  const int ar   = lane & 15;
  const int koff = (lane >= 16) ? 8 : 0;
  const int cc   = lane & 15;
  const int rr   = (lane >> 4) * 8;

  // Async K/V tile staging: per-lane 16B chunk -> per-lane LDS address.
  auto issue_async = [&](int s0, int buf) {
    unsigned lK0 = (unsigned)(uintptr_t)(Kl + buf * 64 * KST + srow * KST + sch * 8);
    unsigned lK1 = lK0 + (unsigned)(32 * KST * 2);
    unsigned lV0 = (unsigned)(uintptr_t)(Vl + buf * 64 * VTST + srow * VTST + sch * 8);
    unsigned lV1 = lV0 + (unsigned)(32 * VTST * 2);
    unsigned o0  = (unsigned)((srow * SEQLEN + s0 + sch * 8) * 2);   // bytes
    unsigned o1  = o0 + (unsigned)(32 * SEQLEN * 2);
    asm volatile("global_load_async_to_lds_b128 %0, %1, %2"
                 :: "v"(lK0), "v"(o0), "s"(kbase) : "memory");
    asm volatile("global_load_async_to_lds_b128 %0, %1, %2"
                 :: "v"(lK1), "v"(o1), "s"(kbase) : "memory");
    asm volatile("global_load_async_to_lds_b128 %0, %1, %2"
                 :: "v"(lV0), "v"(o0), "s"(vbase) : "memory");
    asm volatile("global_load_async_to_lds_b128 %0, %1, %2"
                 :: "v"(lV1), "v"(o1), "s"(vbase) : "memory");
  };

  // Q fragments pinned in VGPRs for the whole S loop (K=64 -> 2 k-steps)
  const __bf16* qrow =
      qb + ((size_t)(l0 + w * 16 + ar) * NBATCH + b) * EMB + h * HDIM;
  v16bf aq0 = cat8(*(const v8bf*)(qrow + koff),      *(const v8bf*)(qrow + 16 + koff));
  v16bf aq1 = cat8(*(const v8bf*)(qrow + 32 + koff), *(const v8bf*)(qrow + 48 + koff));

  v8f oaccT[4];   // out^T: 4 d-tiles x (16 d x 16 l)
  #pragma unroll
  for (int jd = 0; jd < 4; ++jd) oaccT[jd] = zero8();

  __bf16* ptw = Pt + w * 64 * PTST;

  issue_async(0, 0);
  int ibuf = 0;
  for (int s0 = 0; s0 < SEQLEN; s0 += SB) {
    // my async loads done -> barrier -> everyone's tile is visible
    asm volatile("s_wait_asynccnt 0x0" ::: "memory");
    __syncthreads();
    // stream next tile into the other buffer (overlaps the 16 WMMAs below)
    if (s0 + SB < SEQLEN) issue_async(s0 + SB, ibuf ^ 1);

    const __bf16* Kc = Kl + ibuf * 64 * KST;
    const __bf16* Vc = Vl + ibuf * 64 * VTST;

    // Stage 1: P = relu(Q K^T): 4 s-tiles x 2 k-steps = 8 WMMA
    #pragma unroll
    for (int j = 0; j < 4; ++j) {
      v8f p = zero8();
      const __bf16* b0 = Kc + koff * KST + j * 16 + ar;
      const __bf16* b1 = Kc + (32 + koff) * KST + j * 16 + ar;
      v16bf bf0, bf1;
      #pragma unroll
      for (int s = 0; s < 16; ++s) {
        bf0[s] = b0[((s & 7) + (s >> 3) * 16) * KST];
        bf1[s] = b1[((s & 7) + (s >> 3) * 16) * KST];
      }
      p = __builtin_amdgcn_wmma_f32_16x16x32_bf16(false, aq0, false, bf0,
                                                  (short)0, p, false, false);
      p = __builtin_amdgcn_wmma_f32_16x16x32_bf16(false, aq1, false, bf1,
                                                  (short)0, p, false, false);
      // relu (v_med3_f32) + pack 8 rows -> one ds_store_b128 into P^T
      v8bf c;
      #pragma unroll
      for (int r = 0; r < 8; ++r) c[r] = (__bf16)relu1(p[r]);
      *(v8bf*)(ptw + (j * 16 + cc) * PTST + rr) = c;
    }
    // wave-local LDS RAW: split dep counter wait (CDNA5)
    asm volatile("s_wait_dscnt 0x0" ::: "memory");

    // Stage 2: out^T += V^T * P^T: 2 k-steps x 4 d-tiles = 8 WMMA
    #pragma unroll
    for (int ks = 0; ks < 2; ++ks) {
      const __bf16* pr = ptw + (ks * 32 + koff) * PTST + cc;
      v16bf bp;
      #pragma unroll
      for (int s = 0; s < 16; ++s) bp[s] = pr[((s & 7) + (s >> 3) * 16) * PTST];
      #pragma unroll
      for (int jd = 0; jd < 4; ++jd) {
        const __bf16* vrow = Vc + (jd * 16 + ar) * VTST + ks * 32;
        v16bf av = cat8(*(const v8bf*)(vrow + koff),
                        *(const v8bf*)(vrow + 16 + koff));
        oaccT[jd] = __builtin_amdgcn_wmma_f32_16x16x32_bf16(
            false, av, false, bp, (short)0, oaccT[jd], false, false);
      }
    }
    ibuf ^= 1;
  }

  // Epilogue: out^T D-layout: VGPR r -> d row, lane -> l col.
  // 8 consecutive d per lane -> contiguous f32 along E.
  const int l = l0 + w * 16 + cc;
  float* obase = attn + ((size_t)l * NBATCH + b) * EMB + h * HDIM + rr;
  #pragma unroll
  for (int jd = 0; jd < 4; ++jd) {
    float4 lo = {oaccT[jd][0], oaccT[jd][1], oaccT[jd][2], oaccT[jd][3]};
    float4 hi = {oaccT[jd][4], oaccT[jd][5], oaccT[jd][6], oaccT[jd][7]};
    *(float4*)(obase + jd * 16)     = lo;
    *(float4*)(obase + jd * 16 + 4) = hi;
  }
}

// ---------------------------------------------------------------------------
// Gated RMS norm: y = w * x/(rms+eps) * sigmoid(g*x); one float4 per thread
// ---------------------------------------------------------------------------
__global__ __launch_bounds__(256) void gated_rmsnorm(
    const float* __restrict__ x, const float* __restrict__ wgt,
    const float* __restrict__ gate, __bf16* __restrict__ out)
{
  __shared__ float red[8];
  const int n = blockIdx.x, t = threadIdx.x;
  const float4 v = *(const float4*)(x + (size_t)n * EMB + t * 4);

  float ss = v.x * v.x + v.y * v.y + v.z * v.z + v.w * v.w;
  #pragma unroll
  for (int o = 16; o > 0; o >>= 1) ss += __shfl_xor(ss, o, 32);
  if ((t & 31) == 0) red[t >> 5] = ss;
  __syncthreads();
  float tot = 0.f;
  #pragma unroll
  for (int i = 0; i < 8; ++i) tot += red[i];

  float rms = sqrtf(tot * (1.0f / EMB));
  float inv = 1.0f / (rms + EPS);
  const float4 g = *(const float4*)(gate + t * 4);
  const float4 wv = *(const float4*)(wgt + t * 4);
  float y0 = wv.x * v.x * inv / (1.0f + __expf(-g.x * v.x));
  float y1 = wv.y * v.y * inv / (1.0f + __expf(-g.y * v.y));
  float y2 = wv.z * v.z * inv / (1.0f + __expf(-g.z * v.z));
  float y3 = wv.w * v.w * inv / (1.0f + __expf(-g.w * v.w));
  v4bf c = {(__bf16)y0, (__bf16)y1, (__bf16)y2, (__bf16)y3};
  *(v4bf*)(out + (size_t)n * EMB + t * 4) = c;
}

// ---------------------------------------------------------------------------
extern "C" void kernel_launch(void* const* d_in, const int* in_sizes, int n_in,
                              void* d_out, int out_size, void* d_ws, size_t ws_size,
                              hipStream_t stream)
{
  (void)in_sizes; (void)n_in; (void)out_size; (void)ws_size;
  const float* query = (const float*)d_in[0];
  const float* key_  = (const float*)d_in[1];
  const float* value = (const float*)d_in[2];
  const float* Wq = (const float*)d_in[3];  const float* bq = (const float*)d_in[4];
  const float* Wk = (const float*)d_in[5];  const float* bk = (const float*)d_in[6];
  const float* Wv = (const float*)d_in[7];  const float* bv = (const float*)d_in[8];
  const float* Wo = (const float*)d_in[9];  const float* bo = (const float*)d_in[10];
  const float* nw = (const float*)d_in[11]; const float* ng = (const float*)d_in[12];

  // Workspace carve-out (~56 MB total)
  char* ws = (char*)d_ws;
  size_t off = 0;
  auto carve = [&](size_t bytes) {
    char* p = ws + off; off += (bytes + 255) & ~(size_t)255; return p;
  };
  __bf16* qbuf  = (__bf16*)carve((size_t)NTOK * EMB * 2);  // q, scaled, bf16
  __bf16* kTbuf = (__bf16*)carve((size_t)NTOK * EMB * 2);  // k^T per head
  __bf16* vTbuf = (__bf16*)carve((size_t)NTOK * EMB * 2);  // v^T per head
  float*  attn  = (float*) carve((size_t)NTOK * EMB * 4);  // attention out, f32
  __bf16* normb = (__bf16*)carve((size_t)NTOK * EMB * 2);  // gated-rmsnorm
  __bf16* WqT   = (__bf16*)carve((size_t)EMB * EMB * 2);   // weights, [k][n] bf16
  __bf16* WkT   = (__bf16*)carve((size_t)EMB * EMB * 2);
  __bf16* WvT   = (__bf16*)carve((size_t)EMB * EMB * 2);
  __bf16* WoT   = (__bf16*)carve((size_t)EMB * EMB * 2);

  const float scaling = 0.125f;  // HDIM^-0.5
  dim3 gt(EMB / 32, EMB / 32);   // (32, 32) weight transpose
  dim3 gg(EMB / BN, NTOK / BM);  // (8, 32)  GEMM

  // One-time weight transpose+convert (4 MB each f32, L2-resident)
  wtrans_bf16<<<gt, 256, 0, stream>>>(Wq, WqT);
  wtrans_bf16<<<gt, 256, 0, stream>>>(Wk, WkT);
  wtrans_bf16<<<gt, 256, 0, stream>>>(Wv, WvT);
  wtrans_bf16<<<gt, 256, 0, stream>>>(Wo, WoT);

  // Projections (bias fused; Q scaled; K and V stored per-head transposed)
  gemm_wmma<false, 0><<<gg, 256, 0, stream>>>(query, WqT, bq, qbuf,  scaling);
  gemm_wmma<false, 1><<<gg, 256, 0, stream>>>(key_,  WkT, bk, kTbuf, 1.0f);
  gemm_wmma<false, 1><<<gg, 256, 0, stream>>>(value, WvT, bv, vTbuf, 1.0f);

  // Fused relu-attention: grid (L/128, B*H)
  attn_relu_wmma<<<dim3(SEQLEN / 128, NBATCH * NHEAD), 256, 0, stream>>>(
      qbuf, kTbuf, vTbuf, attn);

  // Gated RMS norm (bf16 out)
  gated_rmsnorm<<<NTOK, 256, 0, stream>>>(attn, nw, ng, normb);

  // Output projection -> d_out (f32)
  gemm_wmma<true, 2><<<gg, 256, 0, stream>>>(normb, WoT, bo, d_out, 1.0f);
}
